// GraphAttentionLayer_84275848282631
// MI455X (gfx1250) — matrix operand order repaired and checked by reading, not compiled, not run on hardware
//
#include <hip/hip_runtime.h>

// ---- problem constants (match reference) ----
#define BATCH 8
#define NN 2048
#define DIN 128
#define DOUT 64
#define LEAKY 0.2f
#define LOG2E 1.4426950408889634f

typedef __attribute__((ext_vector_type(16))) __bf16 v16bf;
typedef __attribute__((ext_vector_type(8)))  float  v8f;
typedef __attribute__((ext_vector_type(4)))  float  v4f;
typedef __attribute__((ext_vector_type(4)))  unsigned int v4u;

union AU { unsigned int u[8]; v16bf v; };
union BU { v4u q[2];          v16bf v; };

__device__ __forceinline__ unsigned short f2bf(float x) {
    unsigned int u = __float_as_uint(x);
    u = (u + 0x7fffu + ((u >> 16) & 1u)) >> 16;
    return (unsigned short)u;
}
// round-half-up bf16 pair pack: 2 adds + 1 byte-perm
__device__ __forceinline__ unsigned int pack2bf(float lo, float hi) {
    unsigned int ul = __float_as_uint(lo) + 0x8000u;
    unsigned int uh = __float_as_uint(hi) + 0x8000u;
#if __has_builtin(__builtin_amdgcn_perm)
    return __builtin_amdgcn_perm(uh, ul, 0x07060302u);  // {uh[3],uh[2],ul[3],ul[2]}
#else
    return (ul >> 16) | (uh & 0xffff0000u);
#endif
}
__device__ __forceinline__ float fexp2(float x) {
#if __has_builtin(__builtin_amdgcn_exp2f)
    return __builtin_amdgcn_exp2f(x);
#else
    return exp2f(x);
#endif
}

// ---------------------------------------------------------------------------
// Kernel 1: wh = h @ w (exact f32), emit whT (bf16, [B][DOUT][N]) + s1,s2 (f32)
// ---------------------------------------------------------------------------
__global__ __launch_bounds__(128)
void gat_phase1(const float* __restrict__ h, const float* __restrict__ w,
                const float* __restrict__ a, unsigned short* __restrict__ whT,
                float* __restrict__ s1, float* __restrict__ s2) {
    __shared__ float lh[2 * DIN];
    __shared__ float lwh[2 * DOUT];
    const int t = threadIdx.x;
    const long long gr0 = (long long)blockIdx.x * 2;

    #pragma unroll
    for (int i = 0; i < 2; ++i) {
        int idx = t + i * 128;
        int r = idx >> 7, k = idx & 127;
        lh[idx] = h[(gr0 + r) * DIN + k];
    }
    __syncthreads();

    const int r = t >> 6, f = t & 63;
    float acc = 0.f;
    #pragma unroll 8
    for (int k = 0; k < DIN; ++k)
        acc = fmaf(lh[r * DIN + k], w[k * DOUT + f], acc);

    const long long gr = gr0 + r;              // = b*N + n
    const int b = (int)(gr >> 11);
    const int n = (int)(gr & (NN - 1));
    whT[(((long long)(b * DOUT + f)) << 11) + n] = f2bf(acc);
    lwh[r * DOUT + f] = acc;
    __syncthreads();

    if (t < 2) {
        float a1 = 0.f, a2 = 0.f;
        #pragma unroll 8
        for (int q = 0; q < DOUT; ++q) {
            float v = lwh[t * DOUT + q];
            a1 = fmaf(v, a[q], a1);
            a2 = fmaf(v, a[DOUT + q], a2);
        }
        s1[gr0 + t] = a1;
        s2[gr0 + t] = a2;
    }
}

// ---------------------------------------------------------------------------
// Kernel 2: masked softmax + P@V via bf16 WMMA, bound-shifted (no online max).
//   mhat_i = leaky(s1_i + max_j s2_j) bounds all scores (leaky monotone,
//   rank-1 scores) -> exp2 args <= 0, no rescaling. adj in {0,1}: mask by
//   multiply. Row sums via a 5th WMMA against an all-ones B operand.
// grid = B*(N/16) blocks, 128 threads (4 waves), waves split the j-range.
// ---------------------------------------------------------------------------
__global__ __launch_bounds__(128)
void gat_phase2(const float* __restrict__ adj, const unsigned short* __restrict__ whT,
                const float* __restrict__ s1, const float* __restrict__ s2,
                float* __restrict__ out) {
    __shared__ float sS2I[2 * NN];     // interleaved {s2*log2e, 0.2*s2*log2e}
    __shared__ float sO[4][16 * DOUT];
    __shared__ float sL[4][16];
    __shared__ float sInv[16];
    __shared__ float sRed[4];

    const int t     = threadIdx.x;
    const int wid   = t >> 5;
    const int lane  = t & 31;
    const int hi    = lane & 15;
    const int hiSel = lane >> 4;               // 0: lanes 0-15, 1: lanes 16-31
    const int b     = blockIdx.x >> 7;
    const int i0    = (blockIdx.x & 127) << 4;

    // stage interleaved scaled s2 into LDS; track max(s2*log2e)
    const float* s2b = s2 + ((long long)b << 11);
    float s2mx = -3.0e38f;
    #pragma unroll
    for (int i = 0; i < NN / 128; ++i) {
        float v = s2b[t + i * 128] * LOG2E;
        sS2I[2 * (t + i * 128)]     = v;
        sS2I[2 * (t + i * 128) + 1] = LEAKY * v;
        s2mx = fmaxf(s2mx, v);
    }
    #pragma unroll
    for (int d = 16; d >= 1; d >>= 1) s2mx = fmaxf(s2mx, __shfl_xor(s2mx, d, 32));
    if (lane == 0) sRed[wid] = s2mx;
    __syncthreads();
    s2mx = fmaxf(fmaxf(sRed[0], sRed[1]), fmaxf(sRed[2], sRed[3]));

    const int row = i0 + hi;
    const float s1v  = s1[((long long)b << 11) + row] * LOG2E;
    const float tub  = s1v + s2mx;
    const float mhat = fmaxf(tub, LEAKY * tub);      // row upper bound
    const float s1A  = s1v - mhat;
    const float s1B  = LEAKY * s1v - mhat;
    const v4f s1ABAB = {s1A, s1B, s1A, s1B};

    const int c0  = hiSel << 3;                // A-operand column offset: 0 or 8
    const int jr0 = hiSel << 3;                // B-operand row offset: 0 or 8

    // streaming pointers: advance 128 columns per step
    const float* ap = adj + ((((long long)b << 11) + row) << 11) + (wid << 5) + c0;
    int lofs = ((wid << 5) + c0) * 2;
    const unsigned short* wbb = whT + (((long long)b * DOUT) << 11) + (wid << 5) + jr0;
    const unsigned short* wb0 = wbb + (((long long)( 0 + hi)) << 11);
    const unsigned short* wb1 = wbb + (((long long)(16 + hi)) << 11);
    const unsigned short* wb2 = wbb + (((long long)(32 + hi)) << 11);
    const unsigned short* wb3 = wbb + (((long long)(48 + hi)) << 11);

    BU ONES;
    ONES.q[0] = (v4u){0x3f803f80u, 0x3f803f80u, 0x3f803f80u, 0x3f803f80u};
    ONES.q[1] = ONES.q[0];

    v8f O0 = {}, O1 = {}, O2 = {}, O3 = {}, O4 = {};

    for (int tstep = 0; tstep < 32; ++tstep) {
        // adj tile (16x32 per wave), non-temporal streaming loads
        v4f av[4];
        av[0] = __builtin_nontemporal_load((const v4f*)(ap));
        av[1] = __builtin_nontemporal_load((const v4f*)(ap + 4));
        av[2] = __builtin_nontemporal_load((const v4f*)(ap + 16));
        av[3] = __builtin_nontemporal_load((const v4f*)(ap + 20));
        const v4f* sp = (const v4f*)(sS2I + lofs);

        // p = adj * exp2(max(s1A + s, s1B + t)), packed adds via interleaved LDS
        AU A;
        #pragma unroll
        for (int g = 0; g < 2; ++g) {              // column groups +0, +16
            #pragma unroll
            for (int d = 0; d < 4; ++d) {          // 2 columns per v4f
                v4f q = sp[g * 8 + d];             // {s_j, t_j, s_j1, t_j1}
                v4f x = q + s1ABAB;                // 2x v_pk_add_f32
                float e0 = fexp2(fmaxf(x[0], x[1]));
                float e1 = fexp2(fmaxf(x[2], x[3]));
                float a0 = av[g * 2 + (d >> 1)][(d & 1) * 2];
                float a1 = av[g * 2 + (d >> 1)][(d & 1) * 2 + 1];
                A.u[g * 4 + d] = pack2bf(a0 * e0, a1 * e1);
            }
        }

        // B operands: whT[f][j..j+7] and whT[f][j+16..j+23], two 16B bf16 chunks
        BU B0, B1, B2, B3;
        { const v4u* bp = (const v4u*)wb0; B0.q[0] = bp[0]; B0.q[1] = bp[2]; }
        { const v4u* bp = (const v4u*)wb1; B1.q[0] = bp[0]; B1.q[1] = bp[2]; }
        { const v4u* bp = (const v4u*)wb2; B2.q[0] = bp[0]; B2.q[1] = bp[2]; }
        { const v4u* bp = (const v4u*)wb3; B3.q[0] = bp[0]; B3.q[1] = bp[2]; }

        // O += P @ V ; O4 += P @ ones (row sums)
        O0 = __builtin_amdgcn_wmma_f32_16x16x32_bf16(false, A.v, false, B0.v,   (short)0, O0, false, false);
        O1 = __builtin_amdgcn_wmma_f32_16x16x32_bf16(false, A.v, false, B1.v,   (short)0, O1, false, false);
        O2 = __builtin_amdgcn_wmma_f32_16x16x32_bf16(false, A.v, false, B2.v,   (short)0, O2, false, false);
        O3 = __builtin_amdgcn_wmma_f32_16x16x32_bf16(false, A.v, false, B3.v,   (short)0, O3, false, false);
        O4 = __builtin_amdgcn_wmma_f32_16x16x32_bf16(false, A.v, false, ONES.v, (short)0, O4, false, false);

        ap += 128; lofs += 256;
        wb0 += 128; wb1 += 128; wb2 += 128; wb3 += 128;
    }

    // dump per-wave partials (plain sums -- identical shift mhat per row)
    const int rbase = hiSel << 3;
    if (hi == 0) {                       // lanes 0 and 16: rows 0-7 / 8-15
        #pragma unroll
        for (int rr = 0; rr < 8; ++rr) sL[wid][rbase + rr] = O4[rr];
    }
    #pragma unroll
    for (int rr = 0; rr < 8; ++rr) {
        const int M = rbase + rr;
        sO[wid][M * DOUT +  0 + hi] = O0[rr];
        sO[wid][M * DOUT + 16 + hi] = O1[rr];
        sO[wid][M * DOUT + 32 + hi] = O2[rr];
        sO[wid][M * DOUT + 48 + hi] = O3[rr];
    }
    __syncthreads();

    if (t < 16) {
        float lt = sL[0][t] + sL[1][t] + sL[2][t] + sL[3][t];
        sInv[t] = 1.0f / lt;
    }
    __syncthreads();

    float* ob = out + (((((long long)b << 11) + i0)) << 6);
    #pragma unroll
    for (int i = 0; i < (16 * DOUT) / 128; ++i) {
        const int idx = t + i * 128;
        const int M = idx >> 6;
        float acc = sO[0][idx] + sO[1][idx] + sO[2][idx] + sO[3][idx];
        ob[idx] = acc * sInv[M];
    }
}

// ---------------------------------------------------------------------------
extern "C" void kernel_launch(void* const* d_in, const int* in_sizes, int n_in,
                              void* d_out, int out_size, void* d_ws, size_t ws_size,
                              hipStream_t stream) {
    (void)in_sizes; (void)n_in; (void)out_size; (void)ws_size;
    const float* h   = (const float*)d_in[0];  // (B,N,DIN)
    const float* adj = (const float*)d_in[1];  // (B,N,N)
    const float* w   = (const float*)d_in[2];  // (DIN,DOUT)
    const float* a   = (const float*)d_in[3];  // (2*DOUT,1)
    float* out = (float*)d_out;                // (B,N,DOUT)

    char* ws = (char*)d_ws;
    float* s1 = (float*)ws;                            // B*N f32 = 64 KB
    float* s2 = (float*)(ws + (64 << 10));             // B*N f32 = 64 KB
    unsigned short* whT = (unsigned short*)(ws + (128 << 10)); // B*DOUT*N bf16 = 2 MB

    gat_phase1<<<(BATCH * NN) / 2, 128, 0, stream>>>(h, w, a, whT, s1, s2);
    gat_phase2<<<BATCH * (NN / 16), 128, 0, stream>>>(adj, whT, s1, s2, out);
}
